// GINEncoder_39599598469629
// MI455X (gfx1250) — matrix operand order repaired and checked by reading, not compiled, not run on hardware
//
#include <hip/hip_runtime.h>
#include <hip/hip_bf16.h>
#include <math.h>

typedef __attribute__((ext_vector_type(16))) __bf16 v16bf;
typedef __attribute__((ext_vector_type(8)))  float  v8f;

#define HDIM 128
#define DDIM 64
#define BN_EPS 1e-5f

// ---------------------------------------------------------------------------
// small utility kernels
// ---------------------------------------------------------------------------
__global__ void copy4_kernel(const float4* __restrict__ src, float4* __restrict__ dst,
                             long long n4) {
    long long i = (long long)blockIdx.x * blockDim.x + threadIdx.x;
    long long stride = (long long)gridDim.x * blockDim.x;
    for (; i < n4; i += stride) dst[i] = src[i];
}

__global__ void zero_stats_kernel(float* __restrict__ stats) {
    // zero sum + sumsq (2*HDIM floats)
    int i = threadIdx.x;
    if (i < 2 * HDIM) stats[i] = 0.0f;
}

// transpose + convert weights to bf16: w[K][Nc] -> wt[Nc][K]
__global__ void convw_kernel(const float* __restrict__ w, __bf16* __restrict__ wt,
                             int K, int Nc) {
    int i = blockIdx.x * blockDim.x + threadIdx.x;
    if (i < K * Nc) {
        int k = i / Nc, n = i % Nc;
        wt[n * K + k] = (__bf16)w[i];
    }
}

// ---------------------------------------------------------------------------
// edge scatter: t[dst] += x[src]  (t pre-initialized with x for GIN eps=0)
// one thread = 4 features of one edge -> float4 gather + 4 f32 atomics
// ---------------------------------------------------------------------------
__global__ void edge_agg_kernel(const long long* __restrict__ src,
                                const long long* __restrict__ dst,
                                const float* __restrict__ x,
                                float* __restrict__ t, long long E) {
    long long total  = E * (HDIM / 4);
    long long i      = (long long)blockIdx.x * blockDim.x + threadIdx.x;
    long long stride = (long long)gridDim.x * blockDim.x;
    for (; i < total; i += stride) {
        long long e = i >> 5;               // HDIM/4 == 32 chunks per edge
        int       c = (int)(i & 31) * 4;
        long long s = src[e], d = dst[e];
        float4 v = *(const float4*)(x + s * HDIM + c);
        float* tp = t + d * HDIM + c;
        atomicAdd(tp + 0, v.x);
        atomicAdd(tp + 1, v.y);
        atomicAdd(tp + 2, v.z);
        atomicAdd(tp + 3, v.w);
    }
}

// ---------------------------------------------------------------------------
// WMMA GEMM: [nRows,128] x [128,128] (+bias)
//   RELU_BF16_OUT: apply ReLU, store bf16 (feeds next GEMM's A operand)
//   STATS:         store f32, accumulate per-column sum/sumsq for BatchNorm
// Block: 256 threads = 8 waves; each wave computes a 16x128 row-slab.
// Weight (pre-transposed bf16 wt[n][k]) staged once in LDS (32 KB).
// ---------------------------------------------------------------------------
template <typename InT, bool RELU_BF16_OUT, bool STATS>
__global__ __launch_bounds__(256) void gemm_mlp_kernel(
    const InT* __restrict__ A, const __bf16* __restrict__ Wt,
    const float* __restrict__ bias, float* __restrict__ outF,
    __bf16* __restrict__ outB, float* __restrict__ stats, int nRows) {
    __shared__ __bf16 lds[HDIM * HDIM];  // wt[n][k], 32 KB
    {
        const uint4* s = (const uint4*)Wt;
        uint4*       d = (uint4*)lds;
        #pragma unroll
        for (int i = 0; i < (HDIM * HDIM) / (8 * 256); ++i)
            d[threadIdx.x + i * 256] = s[threadIdx.x + i * 256];
    }
    __syncthreads();

    const int wave    = threadIdx.x >> 5;
    const int lane    = threadIdx.x & 31;
    const int rowBase = blockIdx.x * 128 + wave * 16;
    if (rowBase >= nRows) return;
    const int r  = lane & 15;   // A row within tile / D column
    const int hi = lane >> 4;

    v8f zero = {};
    v8f acc[8];
    #pragma unroll
    for (int t = 0; t < 8; ++t) acc[t] = zero;

    const InT* aRow = A + (size_t)(rowBase + r) * HDIM;
    #pragma unroll
    for (int kt = 0; kt < 4; ++kt) {
        // A fragment (16x32 bf16): lane r, element e holds k = 16*hi + e (contiguous, same as B)
        const int k0 = kt * 32 + hi * 16;
        v16bf afrag;
        #pragma unroll
        for (int e = 0; e < 8; ++e) {
            afrag[e]     = (__bf16)(float)aRow[k0 + e];
            afrag[e + 8] = (__bf16)(float)aRow[k0 + 8 + e];
        }
        #pragma unroll
        for (int t = 0; t < 8; ++t) {
            const int n = t * 16 + r;  // B column for this lane
            const v16bf bfrag =
                *(const v16bf*)&lds[n * HDIM + kt * 32 + hi * 16];
            acc[t] = __builtin_amdgcn_wmma_f32_16x16x32_bf16(
                false, afrag, false, bfrag, (short)0, acc[t], false, false);
        }
    }

    // epilogue: bias (+ReLU / +BN stats), D layout m = r + 8*hi per VGPR rr
    const int col = lane & 15;
    #pragma unroll
    for (int t = 0; t < 8; ++t) {
        const float b = bias[t * 16 + col];
        float s = 0.f, q = 0.f;
        #pragma unroll
        for (int rr = 0; rr < 8; ++rr) {
            float v = acc[t][rr] + b;
            const int m = rowBase + rr + hi * 8;
            if (RELU_BF16_OUT) {
                v = v > 0.f ? v : 0.f;
                outB[(size_t)m * HDIM + t * 16 + col] = (__bf16)v;
            } else {
                outF[(size_t)m * HDIM + t * 16 + col] = v;
                s += v;
                q += v * v;
            }
        }
        if (STATS) {
            s += __shfl_xor(s, 16);
            q += __shfl_xor(q, 16);
            if (hi == 0) {
                atomicAdd(&stats[t * 16 + col], s);
                atomicAdd(&stats[HDIM + t * 16 + col], q);
            }
        }
    }
}

// ---------------------------------------------------------------------------
// BN finalize: scale = g*rsqrt(var+eps); shift = be - mean*scale
// ---------------------------------------------------------------------------
__global__ void bn_stats_kernel(float* __restrict__ stats,
                                const float* __restrict__ g,
                                const float* __restrict__ be, float invN) {
    int c = threadIdx.x;
    if (c >= HDIM) return;
    float mean = stats[c] * invN;
    float var  = stats[HDIM + c] * invN - mean * mean;
    float sc   = g[c] * rsqrtf(var + BN_EPS);
    stats[2 * HDIM + c] = sc;
    stats[3 * HDIM + c] = be[c] - mean * sc;
}

__global__ void bn_apply_kernel(float* __restrict__ h,
                                const float* __restrict__ stats,
                                long long n4) {
    const float* scale = stats + 2 * HDIM;
    const float* shift = stats + 3 * HDIM;
    long long i      = (long long)blockIdx.x * blockDim.x + threadIdx.x;
    long long stride = (long long)gridDim.x * blockDim.x;
    for (; i < n4; i += stride) {
        float4 v = ((float4*)h)[i];
        int c = (int)((i * 4) & (HDIM - 1));
        v.x = fmaxf(v.x * scale[c + 0] + shift[c + 0], 0.f);
        v.y = fmaxf(v.y * scale[c + 1] + shift[c + 1], 0.f);
        v.z = fmaxf(v.z * scale[c + 2] + shift[c + 2], 0.f);
        v.w = fmaxf(v.w * scale[c + 3] + shift[c + 3], 0.f);
        ((float4*)h)[i] = v;
    }
}

// ---------------------------------------------------------------------------
// projection [nRows,128] x [128,64] + bias, fused row L2-normalize
// ---------------------------------------------------------------------------
__global__ __launch_bounds__(256) void proj_norm_kernel(
    const float* __restrict__ A, const __bf16* __restrict__ Wt,
    const float* __restrict__ bias, float* __restrict__ out, int nRows) {
    __shared__ __bf16 lds[DDIM * HDIM];  // 16 KB
    {
        const uint4* s = (const uint4*)Wt;
        uint4*       d = (uint4*)lds;
        #pragma unroll
        for (int i = 0; i < (DDIM * HDIM) / (8 * 256); ++i)
            d[threadIdx.x + i * 256] = s[threadIdx.x + i * 256];
    }
    __syncthreads();

    const int wave    = threadIdx.x >> 5;
    const int lane    = threadIdx.x & 31;
    const int rowBase = blockIdx.x * 128 + wave * 16;
    if (rowBase >= nRows) return;
    const int r  = lane & 15;
    const int hi = lane >> 4;

    v8f zero = {};
    v8f acc[4];
    #pragma unroll
    for (int t = 0; t < 4; ++t) acc[t] = zero;

    const float* aRow = A + (size_t)(rowBase + r) * HDIM;
    #pragma unroll
    for (int kt = 0; kt < 4; ++kt) {
        const int k0 = kt * 32 + hi * 16;
        v16bf afrag;
        #pragma unroll
        for (int e = 0; e < 8; ++e) {
            afrag[e]     = (__bf16)aRow[k0 + e];
            afrag[e + 8] = (__bf16)aRow[k0 + 8 + e];
        }
        #pragma unroll
        for (int t = 0; t < 4; ++t) {
            const int n = t * 16 + r;
            const v16bf bfrag =
                *(const v16bf*)&lds[n * HDIM + kt * 32 + hi * 16];
            acc[t] = __builtin_amdgcn_wmma_f32_16x16x32_bf16(
                false, afrag, false, bfrag, (short)0, acc[t], false, false);
        }
    }

    // bias + per-row sum of squares (rows r+8*hi live in one 16-lane group)
    const int col = lane & 15;
    float vals[4][8];
    float sq[8];
    #pragma unroll
    for (int rr = 0; rr < 8; ++rr) sq[rr] = 0.f;
    #pragma unroll
    for (int t = 0; t < 4; ++t) {
        const float b = bias[t * 16 + col];
        #pragma unroll
        for (int rr = 0; rr < 8; ++rr) {
            float v = acc[t][rr] + b;
            vals[t][rr] = v;
            sq[rr] += v * v;
        }
    }
    #pragma unroll
    for (int rr = 0; rr < 8; ++rr) {
        float s = sq[rr];
        s += __shfl_xor(s, 1);
        s += __shfl_xor(s, 2);
        s += __shfl_xor(s, 4);
        s += __shfl_xor(s, 8);
        sq[rr] = 1.0f / fmaxf(sqrtf(s), 1e-12f);
    }
    #pragma unroll
    for (int t = 0; t < 4; ++t)
        #pragma unroll
        for (int rr = 0; rr < 8; ++rr) {
            const int m = rowBase + rr + hi * 8;
            out[(size_t)m * DDIM + t * 16 + col] = vals[t][rr] * sq[rr];
        }
}

// ---------------------------------------------------------------------------
// host-side orchestration
// ---------------------------------------------------------------------------
extern "C" void kernel_launch(void* const* d_in, const int* in_sizes, int n_in,
                              void* d_out, int out_size, void* d_ws,
                              size_t ws_size, hipStream_t stream) {
    const float*     x   = (const float*)d_in[0];
    const long long* ei  = (const long long*)d_in[1];
    const float *w1_0 = (const float*)d_in[2],  *b1_0 = (const float*)d_in[3];
    const float *w2_0 = (const float*)d_in[4],  *b2_0 = (const float*)d_in[5];
    const float *g0   = (const float*)d_in[6],  *be0  = (const float*)d_in[7];
    const float *w1_1 = (const float*)d_in[8],  *b1_1 = (const float*)d_in[9];
    const float *w2_1 = (const float*)d_in[10], *b2_1 = (const float*)d_in[11];
    const float *g1   = (const float*)d_in[12], *be1  = (const float*)d_in[13];
    const float *wp   = (const float*)d_in[14], *bp   = (const float*)d_in[15];

    const int       N = in_sizes[0] / HDIM;
    const long long E = (long long)in_sizes[1] / 2;
    const long long* src = ei;
    const long long* dst = ei + E;

    // workspace layout
    const size_t NH4 = (size_t)N * HDIM * sizeof(float);
    const size_t NH2 = (size_t)N * HDIM * sizeof(__bf16);
    char* ws = (char*)d_ws;
    float*  bufA  = (float*)(ws);                 // x + agg
    float*  bufB  = (float*)(ws + NH4);           // MLP out / BN out
    __bf16* bufU  = (__bf16*)(ws + 2 * NH4);      // hidden (bf16)
    __bf16* wt1_0 = (__bf16*)(ws + 2 * NH4 + NH2);
    __bf16* wt2_0 = wt1_0 + HDIM * HDIM;
    __bf16* wt1_1 = wt2_0 + HDIM * HDIM;
    __bf16* wt2_1 = wt1_1 + HDIM * HDIM;
    __bf16* wtp   = wt2_1 + HDIM * HDIM;
    float*  stats = (float*)((char*)(wtp + DDIM * HDIM) + 128);
    float*  outF  = (float*)d_out;

    const int gemmBlocks = (N + 127) / 128;
    const long long n4   = (long long)N * HDIM / 4;
    const int copyBlocks = 2048, edgeBlocks = 4096, ewBlocks = 2048;
    const float invN = 1.0f / (float)N;

    // weight transpose+convert to bf16 (tiny, once per call)
    convw_kernel<<<(HDIM * HDIM + 255) / 256, 256, 0, stream>>>(w1_0, wt1_0, HDIM, HDIM);
    convw_kernel<<<(HDIM * HDIM + 255) / 256, 256, 0, stream>>>(w2_0, wt2_0, HDIM, HDIM);
    convw_kernel<<<(HDIM * HDIM + 255) / 256, 256, 0, stream>>>(w1_1, wt1_1, HDIM, HDIM);
    convw_kernel<<<(HDIM * HDIM + 255) / 256, 256, 0, stream>>>(w2_1, wt2_1, HDIM, HDIM);
    convw_kernel<<<(HDIM * DDIM + 255) / 256, 256, 0, stream>>>(wp, wtp, HDIM, DDIM);

    // ---- layer 0 ----
    copy4_kernel<<<copyBlocks, 256, 0, stream>>>((const float4*)x, (float4*)bufA, n4);
    edge_agg_kernel<<<edgeBlocks, 256, 0, stream>>>(src, dst, x, bufA, E);
    zero_stats_kernel<<<1, 256, 0, stream>>>(stats);
    gemm_mlp_kernel<float, true, false><<<gemmBlocks, 256, 0, stream>>>(
        bufA, wt1_0, b1_0, nullptr, bufU, nullptr, N);
    gemm_mlp_kernel<__bf16, false, true><<<gemmBlocks, 256, 0, stream>>>(
        bufU, wt2_0, b2_0, bufB, nullptr, stats, N);
    bn_stats_kernel<<<1, HDIM, 0, stream>>>(stats, g0, be0, invN);
    bn_apply_kernel<<<ewBlocks, 256, 0, stream>>>(bufB, stats, n4);

    // ---- layer 1 ----
    copy4_kernel<<<copyBlocks, 256, 0, stream>>>((const float4*)bufB, (float4*)bufA, n4);
    edge_agg_kernel<<<edgeBlocks, 256, 0, stream>>>(src, dst, bufB, bufA, E);
    zero_stats_kernel<<<1, 256, 0, stream>>>(stats);
    gemm_mlp_kernel<float, true, false><<<gemmBlocks, 256, 0, stream>>>(
        bufA, wt1_1, b1_1, nullptr, bufU, nullptr, N);
    gemm_mlp_kernel<__bf16, false, true><<<gemmBlocks, 256, 0, stream>>>(
        bufU, wt2_1, b2_1, bufB, nullptr, stats, N);
    bn_stats_kernel<<<1, HDIM, 0, stream>>>(stats, g1, be1, invN);
    bn_apply_kernel<<<ewBlocks, 256, 0, stream>>>(bufB, stats, n4);

    // ---- projection + L2 normalize ----
    proj_norm_kernel<<<gemmBlocks, 256, 0, stream>>>(bufB, wtp, bp, outF, N);
}
